// NonLocalTokenEnhancement_11166914970117
// MI455X (gfx1250) — matrix operand order repaired and checked by reading, not compiled, not run on hardware
//
#include <hip/hip_runtime.h>
#include <hip/hip_bf16.h>

typedef __attribute__((ext_vector_type(16))) __bf16 v16bf;
typedef __attribute__((ext_vector_type(8)))  float  v8f;
typedef __attribute__((ext_vector_type(4)))  float  v4f;

#define DEV static __device__ __forceinline__

// ---------------- wave32 reductions ----------------
DEV float wave_sum(float v) {
#pragma unroll
  for (int o = 16; o > 0; o >>= 1) v += __shfl_xor(v, o, 32);
  return v;
}
DEV float wave_max(float v) {
#pragma unroll
  for (int o = 16; o > 0; o >>= 1) v = fmaxf(v, __shfl_xor(v, o, 32));
  return v;
}

DEV v4f zero4() { v4f z = {0.f, 0.f, 0.f, 0.f}; return z; }

// ---- fragment loaders (ISA 7.12.2 wave32 layouts) ----
// A row fragment: ap = row base; halves[0..7]=K(k0+half*8..+7), halves[8..15]=+16
template <int ASTRIDE, bool AHALF>
DEV v16bf load_a(const float* __restrict__ ap, int k0, int half) {
  v16bf a;
  if (ASTRIDE == 1) {
    const float* pa = ap + k0 + half * 8;
    v4f f0 = *(const v4f*)(pa);
    v4f f1 = *(const v4f*)(pa + 4);
    v4f f2 = AHALF ? zero4() : *(const v4f*)(pa + 16);
    v4f f3 = AHALF ? zero4() : *(const v4f*)(pa + 20);
#pragma unroll
    for (int t = 0; t < 4; ++t) {
      a[t] = (__bf16)f0[t]; a[t + 4]  = (__bf16)f1[t];
      a[t + 8] = (__bf16)f2[t]; a[t + 12] = (__bf16)f3[t];
    }
  } else {
    const float* pa = ap + (k0 + half * 8) * ASTRIDE;
#pragma unroll
    for (int t = 0; t < 8; ++t) {
      a[t]     = (__bf16)pa[t * ASTRIDE];
      a[t + 8] = AHALF ? (__bf16)0.f : (__bf16)pa[(t + 16) * ASTRIDE];
    }
  }
  return a;
}

// B col fragment: 16 K-values of one column; [0..7]=K(k0+half*8..+7), [8..15]=+16
template <bool BKN>
DEV v16bf load_b(const float* __restrict__ bp, int ldb, int k0, int half) {
  v16bf b;
  int kb = k0 + half * 8;
  if (!BKN) {                       // (Nout x K) row-major: column = contiguous row
    const float* pb = bp + kb;
    v4f g0 = *(const v4f*)(pb);
    v4f g1 = *(const v4f*)(pb + 4);
    v4f g2 = *(const v4f*)(pb + 16);
    v4f g3 = *(const v4f*)(pb + 20);
#pragma unroll
    for (int t = 0; t < 4; ++t) {
      b[t] = (__bf16)g0[t]; b[t + 4]  = (__bf16)g1[t];
      b[t + 8] = (__bf16)g2[t]; b[t + 12] = (__bf16)g3[t];
    }
  } else {                          // (K x Nout) row-major: strided by ldb
    const float* pb = bp + kb * ldb;
#pragma unroll
    for (int t = 0; t < 8; ++t) { b[t] = (__bf16)pb[t * ldb]; b[t + 8] = (__bf16)pb[(t + 16) * ldb]; }
  }
  return b;
}

DEV v8f wmma_bf16(const v16bf& a, const v16bf& b, const v8f& c) {
  return __builtin_amdgcn_wmma_f32_16x16x32_bf16(false, a, false, b, (short)0, c,
                                                 false, false);
}

// ---------------- batched WMMA GEMM: C = scale*(A @ B) + bias ----------------
// One wave computes a 32 x (16*TN) output block (2 x TN register tiles):
// 2 A-frags + TN B-frags -> 2*TN wmma per K-step. M % 32 == 0, Nout % (16*TN) == 0,
// K % 32 == 0 (AHALF: logical K=16, launch K=32; A upper half compile-time zero,
// B loaded unconditionally from in-bounds garbage that is multiplied by 0).
template <bool BKN, int ASTRIDE, bool AHALF, int TN>
__global__ __launch_bounds__(32)
void k_gemm(const float* __restrict__ A, int lda, long long a_ys, long long a_zs,
            const float* __restrict__ Bm, int ldb, long long b_ys, long long b_zs,
            const float* __restrict__ bias,
            float* __restrict__ Cm, int ldc, long long c_ys, long long c_zs,
            int Nout, int K, float scale)
{
  int ntn = Nout / (16 * TN);
  int ti = blockIdx.x / ntn;
  int tj = blockIdx.x - ti * ntn;
  const float* Ab = A  + (long long)blockIdx.y * a_ys + (long long)blockIdx.z * a_zs;
  const float* Bb = Bm + (long long)blockIdx.y * b_ys + (long long)blockIdx.z * b_zs;
  float*       Cb = Cm + (long long)blockIdx.y * c_ys + (long long)blockIdx.z * c_zs;

  int lane = threadIdx.x & 31;
  int half = lane >> 4;
  int l15  = lane & 15;
  int row0 = ti * 32;
  int col0 = tj * 16 * TN;

  const float* ap0 = Ab + (row0 + l15) * lda;
  const float* ap1 = ap0 + 16 * lda;
  const float* bp0 = BKN ? (Bb + col0 + l15) : (Bb + (col0 + l15) * ldb);
  const float* bp1 = BKN ? (bp0 + 16)        : (bp0 + 16 * ldb);

  v8f z = {0.f, 0.f, 0.f, 0.f, 0.f, 0.f, 0.f, 0.f};
  v8f acc00 = z, acc01 = z, acc10 = z, acc11 = z;

  for (int k0 = 0; k0 < K; k0 += 32) {
    v16bf a0 = load_a<ASTRIDE, AHALF>(ap0, k0, half);
    v16bf a1 = load_a<ASTRIDE, AHALF>(ap1, k0, half);
    v16bf b0 = load_b<BKN>(bp0, ldb, k0, half);
    acc00 = wmma_bf16(a0, b0, acc00);
    acc10 = wmma_bf16(a1, b0, acc10);
    if (TN == 2) {
      v16bf b1 = load_b<BKN>(bp1, ldb, k0, half);
      acc01 = wmma_bf16(a0, b1, acc01);
      acc11 = wmma_bf16(a1, b1, acc11);
    }
  }

  int rbase = row0 + (half << 3);
#pragma unroll
  for (int n = 0; n < TN; ++n) {
    int col = col0 + n * 16 + l15;
    float bv = bias ? bias[col] : 0.f;
    const v8f& lo = n ? acc01 : acc00;
    const v8f& hi = n ? acc11 : acc10;
#pragma unroll
    for (int r = 0; r < 8; ++r) {
      Cb[(rbase + r) * ldc + col]        = lo[r] * scale + bv;
      Cb[(rbase + 16 + r) * ldc + col]   = hi[r] * scale + bv;
    }
  }
}

// ---------------- layout / elementwise kernels ----------------
__global__ void k_to_tokens(const float* __restrict__ x, float* __restrict__ tok,
                            int B, int C, int HW) {
  int idx = blockIdx.x * blockDim.x + threadIdx.x;
  if (idx >= B * C * HW) return;
  int c = idx % C;
  int n = (idx / C) % HW;
  int b = idx / (C * HW);
  tok[idx] = x[((size_t)b * C + c) * HW + n];   // idx == ((b*HW+n)*C + c)
}

// softmax over L elements at given stride; row r: base = (r/inner)*outer_stride + (r%inner)
__global__ __launch_bounds__(32)
void k_softmax_rows(float* __restrict__ data, int L, int stride, int inner,
                    long long outer_stride) {
  long long row = blockIdx.x;
  float* p = data + (row / inner) * outer_stride + (row % inner);
  int lane = threadIdx.x;
  float mx = -3.4e38f;
  for (int j = lane; j < L; j += 32) mx = fmaxf(mx, p[j * stride]);
  mx = wave_max(mx);
  float s = 0.f;
  for (int j = lane; j < L; j += 32) {
    float e = __expf(p[j * stride] - mx);
    p[j * stride] = e;
    s += e;
  }
  float inv = 1.f / wave_sum(s);
  for (int j = lane; j < L; j += 32) p[j * stride] *= inv;
}

// out = LN(a (+ res)) * w + b, optional exact GELU; row length fixed 128, one wave per row
__global__ __launch_bounds__(32)
void k_add_ln(const float* __restrict__ a, const float* __restrict__ res,
              const float* __restrict__ w, const float* __restrict__ bias,
              float* __restrict__ out, int do_gelu) {
  const int C = 128;
  size_t row = blockIdx.x;
  int lane = threadIdx.x;
  const float* pa = a + row * C;
  const float* pr = res ? res + row * C : nullptr;
  float x[4];
  float s = 0.f;
#pragma unroll
  for (int t = 0; t < 4; ++t) {
    int c = lane + t * 32;
    x[t] = pa[c] + (pr ? pr[c] : 0.f);
    s += x[t];
  }
  float mean = wave_sum(s) * (1.f / 128.f);
  float v = 0.f;
#pragma unroll
  for (int t = 0; t < 4; ++t) { float d = x[t] - mean; v += d * d; }
  float inv = rsqrtf(wave_sum(v) * (1.f / 128.f) + 1e-5f);
#pragma unroll
  for (int t = 0; t < 4; ++t) {
    int c = lane + t * 32;
    float y = (x[t] - mean) * inv * w[c] + bias[c];
    if (do_gelu) y = 0.5f * y * (1.f + erff(y * 0.70710678118654752f));
    out[row * C + c] = y;
  }
}

// Fused GCN edge: ew[b,i,j,h] = GELU(LN(xa[b,i,:]+xb[b,j,:])) . e2_w[h,:] + e2_b[h]
// Never materializes the (B,N,N,C) tensor. 8 waves/block, one (i,j) pair per wave.
__global__ __launch_bounds__(256)
void k_gcn_edge(const float* __restrict__ xa, const float* __restrict__ xb,
                const float* __restrict__ eln_w, const float* __restrict__ eln_b,
                const float* __restrict__ e2w, const float* __restrict__ e2b,
                float* __restrict__ ew, int N) {
  const int C = 128;
  int b = blockIdx.y;
  int wid = threadIdx.x >> 5;
  int lane = threadIdx.x & 31;
  long long pair = (long long)blockIdx.x * 8 + wid;
  if (pair >= (long long)N * N) return;
  int i = (int)(pair / N), j = (int)(pair % N);
  const float* pa = xa + ((size_t)b * N + i) * C;   // e1_b already folded into xa
  const float* pb = xb + ((size_t)b * N + j) * C;
  float e0[4];
  float s = 0.f;
#pragma unroll
  for (int t = 0; t < 4; ++t) {
    int c = lane + t * 32;
    e0[t] = pa[c] + pb[c];
    s += e0[t];
  }
  float mean = wave_sum(s) * (1.f / 128.f);
  float v = 0.f;
#pragma unroll
  for (int t = 0; t < 4; ++t) { float d = e0[t] - mean; v += d * d; }
  float inv = rsqrtf(wave_sum(v) * (1.f / 128.f) + 1e-5f);
  float y[4];
#pragma unroll
  for (int t = 0; t < 4; ++t) {
    int c = lane + t * 32;
    float z = (e0[t] - mean) * inv * eln_w[c] + eln_b[c];
    y[t] = 0.5f * z * (1.f + erff(z * 0.70710678118654752f));
  }
  float* pw = ew + (((size_t)b * N + i) * N + j) * 4;
#pragma unroll
  for (int h = 0; h < 4; ++h) {
    float acc = 0.f;
#pragma unroll
    for (int t = 0; t < 4; ++t) acc += y[t] * e2w[h * C + lane + t * 32];
    acc = wave_sum(acc);
    if (lane == 0) pw[h] = acc + e2b[h];
  }
}

__global__ void k_concat(const float* __restrict__ a, const float* __restrict__ g,
                         float* __restrict__ out, int rows) {
  int idx = blockIdx.x * blockDim.x + threadIdx.x;
  if (idx >= rows * 256) return;
  int c = idx & 255;
  int r = idx >> 8;
  out[idx] = (c < 128) ? a[(size_t)r * 128 + c] : g[(size_t)r * 128 + (c - 128)];
}

__global__ void k_final(const float* __restrict__ tok, const float* __restrict__ f,
                        float* __restrict__ out, int B, int C, int HW) {
  int idx = blockIdx.x * blockDim.x + threadIdx.x;
  if (idx >= B * C * HW) return;                    // idx == (b*C+c)*HW + n
  int n = idx % HW;
  int c = (idx / HW) % C;
  int b = idx / (HW * C);
  size_t t = ((size_t)b * HW + n) * C + c;
  out[idx] = tok[t] + f[t];
}

// ---------------- host orchestration ----------------
extern "C" void kernel_launch(void* const* d_in, const int* in_sizes, int n_in,
                              void* d_out, int out_size, void* d_ws, size_t ws_size,
                              hipStream_t stream) {
  (void)in_sizes; (void)n_in; (void)out_size; (void)ws_size;
  const int B = 2, C = 128, H = 24, W = 24;
  const int HW = H * W, N = HW, M = B * N;
  const int NH = 8;                   // MHA heads (head dim 16)
  const int NG = 4, HG = 32;          // GCN heads

  auto in = [&](int i) { return (const float*)d_in[i]; };
  const float* x         = in(0);
  const float* mha_in_w  = in(1);  const float* mha_in_b  = in(2);
  const float* mha_out_w = in(3);  const float* mha_out_b = in(4);
  const float* n1_w = in(29); const float* n1_b = in(30);
  const float* n2_w = in(31); const float* n2_b = in(32);
  const float* f1_w = in(33); const float* f1_b = in(34);
  const float* fln_w = in(35); const float* fln_b = in(36);
  const float* f2_w = in(37); const float* f2_b = in(38);

  float* ws = (float*)d_ws;
  size_t off = 0;
  auto alloc = [&](size_t n) { float* p = ws + off; off += n; return p; };
  float* tokens   = alloc((size_t)M * C);
  float* qkv      = alloc((size_t)M * 3 * C);
  float* scores   = alloc((size_t)B * NH * N * N);
  float* attn_tmp = alloc((size_t)M * C);
  float* proj     = alloc((size_t)M * C);
  float* attn     = alloc((size_t)M * C);
  float* vbuf     = alloc((size_t)M * C);
  float* xa       = alloc((size_t)M * C);
  float* xb       = alloc((size_t)M * C);
  float* adj      = alloc((size_t)B * N * N * NG);
  float* gout     = alloc((size_t)M * C);
  float* gtmp     = alloc((size_t)M * C);
  float* gA       = alloc((size_t)M * C);
  float* gB       = alloc((size_t)M * C);
  float* gln      = alloc((size_t)M * C);
  float* cat      = alloc((size_t)M * 2 * C);
  float* fbuf     = alloc((size_t)M * C);
  float* fact     = alloc((size_t)M * C);
  float* fout     = alloc((size_t)M * C);

  // mode: 0 = B(N,K) weight, 32x32/wave; 1 = scores (logical K=16), 32x32/wave;
  //       2 = B(K,N), 32x16/wave (Nout=16);  3 = B(K,N), A stride 4, 32x32/wave
  auto gemm = [&](int mode,
                  const float* A, int lda, long long a_ys, long long a_zs,
                  const float* Bm, int ldb, long long b_ys, long long b_zs,
                  const float* bias, float* Cm, int ldc, long long c_ys, long long c_zs,
                  int Mm, int Nout, int K, int ny, int nz, float scale) {
    int tn = (mode == 2) ? 1 : 2;
    dim3 g((Mm / 32) * (Nout / (16 * tn)), ny, nz), blk(32);
    switch (mode) {
      case 0: k_gemm<false, 1, false, 2><<<g, blk, 0, stream>>>(A, lda, a_ys, a_zs, Bm,
                ldb, b_ys, b_zs, bias, Cm, ldc, c_ys, c_zs, Nout, K, scale); break;
      case 1: k_gemm<false, 1, true , 2><<<g, blk, 0, stream>>>(A, lda, a_ys, a_zs, Bm,
                ldb, b_ys, b_zs, bias, Cm, ldc, c_ys, c_zs, Nout, K, scale); break;
      case 2: k_gemm<true , 1, false, 1><<<g, blk, 0, stream>>>(A, lda, a_ys, a_zs, Bm,
                ldb, b_ys, b_zs, bias, Cm, ldc, c_ys, c_zs, Nout, K, scale); break;
      default:k_gemm<true , 4, false, 2><<<g, blk, 0, stream>>>(A, lda, a_ys, a_zs, Bm,
                ldb, b_ys, b_zs, bias, Cm, ldc, c_ys, c_zs, Nout, K, scale); break;
    }
  };

  // ---- tokens = transpose(x) ----
  k_to_tokens<<<(B * C * HW + 255) / 256, 256, 0, stream>>>(x, tokens, B, C, HW);

  // ---- MHA ----
  gemm(0, tokens, C, 0, 0, mha_in_w, C, 0, 0, mha_in_b,
       qkv, 3 * C, 0, 0, M, 3 * C, C, 1, 1, 1.f);
  // scores[b,h] = Q_h @ K_h^T * 0.25  (grid y=head, z=batch; logical K=16, launch K=32)
  gemm(1, qkv, 3 * C, 16, (long long)N * 3 * C,
       qkv + C, 3 * C, 16, (long long)N * 3 * C, nullptr,
       scores, N, (long long)N * N, (long long)NH * N * N,
       N, N, 32, NH, B, 0.25f);
  k_softmax_rows<<<B * NH * N, 32, 0, stream>>>(scores, N, 1, 1, (long long)N);
  // o = A @ V_h   (Nout = 16 -> 32x16 per wave)
  gemm(2, scores, N, (long long)N * N, (long long)NH * N * N,
       qkv + 2 * C, 3 * C, 16, (long long)N * 3 * C, nullptr,
       attn_tmp, C, 16, (long long)N * C,
       N, 16, N, NH, B, 1.f);
  gemm(0, attn_tmp, C, 0, 0, mha_out_w, C, 0, 0, mha_out_b,
       proj, C, 0, 0, M, C, C, 1, 1, 1.f);
  k_add_ln<<<M, 32, 0, stream>>>(proj, tokens, n1_w, n1_b, attn, 0);

  // ---- GCN blocks ----
  auto run_gcn = [&](const float* xin, float* xout, int gi) {
    int base = 5 + gi * 12;
    const float* v_w   = in(base + 0); const float* v_b   = in(base + 1);
    const float* e1_w  = in(base + 2); const float* e1_b  = in(base + 3);
    const float* eln_w = in(base + 4); const float* eln_b = in(base + 5);
    const float* e2_w  = in(base + 6); const float* e2_b  = in(base + 7);
    const float* o_w   = in(base + 8); const float* o_b   = in(base + 9);
    const float* n_w   = in(base + 10); const float* n_b  = in(base + 11);

    gemm(0, xin, C, 0, 0, v_w, C, 0, 0, v_b, vbuf, C, 0, 0, M, C, C, 1, 1, 1.f);
    gemm(0, xin, C, 0, 0, e1_w, 2 * C, 0, 0, e1_b, xa, C, 0, 0, M, C, C, 1, 1, 1.f);
    gemm(0, xin, C, 0, 0, e1_w + C, 2 * C, 0, 0, nullptr, xb, C, 0, 0, M, C, C, 1, 1, 1.f);

    dim3 eg(N * N / 8, B);
    k_gcn_edge<<<eg, 256, 0, stream>>>(xa, xb, eln_w, eln_b, e2_w, e2_b, adj, N);
    // softmax over j for each (b,i,h): inner=NG, stride=NG
    k_softmax_rows<<<B * N * NG, 32, 0, stream>>>(adj, N, NG, NG, (long long)N * NG);
    // out[b,:,h*32:(h+1)*32] = adj[b,:,:,h] @ vbuf[b,:,h*32:(h+1)*32]  (y=h, z=b)
    gemm(3, adj, N * NG, 1, (long long)N * N * NG,
         vbuf, C, HG, (long long)N * C, nullptr,
         gout, C, HG, (long long)N * C,
         N, HG, N, NG, B, 1.f);
    gemm(0, gout, C, 0, 0, o_w, C, 0, 0, o_b, gtmp, C, 0, 0, M, C, C, 1, 1, 1.f);
    k_add_ln<<<M, 32, 0, stream>>>(gtmp, xin, n_w, n_b, xout, 0);
  };
  run_gcn(attn, gA, 0);
  run_gcn(gA, gB, 1);

  // ---- fusion head ----
  k_add_ln<<<M, 32, 0, stream>>>(gB, nullptr, n2_w, n2_b, gln, 0);
  k_concat<<<(M * 2 * C + 255) / 256, 256, 0, stream>>>(attn, gln, cat, M);
  gemm(0, cat, 2 * C, 0, 0, f1_w, 2 * C, 0, 0, f1_b,
       fbuf, C, 0, 0, M, C, 2 * C, 1, 1, 1.f);
  k_add_ln<<<M, 32, 0, stream>>>(fbuf, nullptr, fln_w, fln_b, fact, 1);
  gemm(0, fact, C, 0, 0, f2_w, C, 0, 0, f2_b,
       fout, C, 0, 0, M, C, C, 1, 1, 1.f);
  k_final<<<(B * C * HW + 255) / 256, 256, 0, stream>>>(tokens, fout, (float*)d_out, B, C, HW);
}
